// EnhancedHBitLinear_5600637354691
// MI455X (gfx1250) — 8-bit matrix operand order repaired and checked by reading, compiled, not run on hardware
//
#include <hip/hip_runtime.h>
#include <stdint.h>

#define IN_F   4096
#define OUT_F  16384
#define ACT_QB 7.0f
#define EPSF   1e-5f
#define RED_BLOCKS 2048

typedef __attribute__((ext_vector_type(16))) int   v16i;
typedef __attribute__((ext_vector_type(8)))  float v8f;

typedef __attribute__((__vector_size__(16))) int v4i_t;
typedef __attribute__((__vector_size__(8)))  int v2i_t;
typedef __attribute__((address_space(1))) v4i_t g_v4i;
typedef __attribute__((address_space(3))) v4i_t l_v4i;
typedef __attribute__((address_space(1))) v2i_t g_v2i;
typedef __attribute__((address_space(3))) v2i_t l_v2i;
typedef __attribute__((address_space(3))) void as3_void;

__device__ __forceinline__ void async_b128(void* lds_dst, const void* gsrc) {
#if __has_builtin(__builtin_amdgcn_global_load_async_to_lds_b128)
  __builtin_amdgcn_global_load_async_to_lds_b128((g_v4i*)gsrc, (l_v4i*)lds_dst, 0, 0);
#else
  unsigned l = (unsigned)(unsigned long long)(as3_void*)lds_dst;
  asm volatile("global_load_async_to_lds_b128 %0, %1, off" :: "v"(l), "v"(gsrc) : "memory");
#endif
}
__device__ __forceinline__ void async_b64(void* lds_dst, const void* gsrc) {
#if __has_builtin(__builtin_amdgcn_global_load_async_to_lds_b64)
  __builtin_amdgcn_global_load_async_to_lds_b64((g_v2i*)gsrc, (l_v2i*)lds_dst, 0, 0);
#else
  unsigned l = (unsigned)(unsigned long long)(as3_void*)lds_dst;
  asm volatile("global_load_async_to_lds_b64 %0, %1, off" :: "v"(l), "v"(gsrc) : "memory");
#endif
}
__device__ __forceinline__ void wait_async_le8() {
#if __has_builtin(__builtin_amdgcn_s_wait_asynccnt)
  __builtin_amdgcn_s_wait_asynccnt(8);
#else
  asm volatile("s_wait_asynccnt 0x8" ::: "memory");
#endif
}
__device__ __forceinline__ void wait_async_le0() {
#if __has_builtin(__builtin_amdgcn_s_wait_asynccnt)
  __builtin_amdgcn_s_wait_asynccnt(0);
#else
  asm volatile("s_wait_asynccnt 0x0" ::: "memory");
#endif
}

static __device__ __forceinline__ unsigned fp8_e4m3_from_small_int(float v) {
  unsigned u = __float_as_uint(v);
  unsigned b = ((u >> 24) & 0x80u) | ((((u >> 23) & 0xffu) - 120u) << 3) | ((u >> 20) & 0x7u);
  return (v == 0.0f) ? 0u : (b & 0xffu);
}

__global__ __launch_bounds__(256) void wabs_partial_kernel(
    const float* __restrict__ w, float* __restrict__ partials, int n4) {
  const float4* w4 = (const float4*)w;
  float s = 0.0f;
  for (int i = blockIdx.x * 256 + threadIdx.x; i < n4; i += gridDim.x * 256) {
    float4 v = w4[i];
    s += fabsf(v.x) + fabsf(v.y) + fabsf(v.z) + fabsf(v.w);
  }
  __shared__ float red[256];
  red[threadIdx.x] = s;
  __syncthreads();
  for (int off = 128; off > 0; off >>= 1) {
    if (threadIdx.x < off) red[threadIdx.x] += red[threadIdx.x + off];
    __syncthreads();
  }
  if (threadIdx.x == 0) partials[blockIdx.x] = red[0];
}

__global__ __launch_bounds__(256) void wabs_final_kernel(
    const float* __restrict__ partials, float* __restrict__ wsum) {
  float s = 0.0f;
  for (int i = threadIdx.x; i < RED_BLOCKS; i += 256) s += partials[i];
  __shared__ float red[256];
  red[threadIdx.x] = s;
  __syncthreads();
  for (int off = 128; off > 0; off >>= 1) {
    if (threadIdx.x < off) red[threadIdx.x] += red[threadIdx.x + off];
    __syncthreads();
  }
  if (threadIdx.x == 0) wsum[0] = red[0];
}

__global__ __launch_bounds__(256) void wquant_kernel(
    const float* __restrict__ w, const float* __restrict__ wsum,
    unsigned char* __restrict__ wq) {
  const float scale = wsum[0] * (1.0f / ((float)IN_F * (float)OUT_F)) + EPSF;
  const float inv = 1.0f / scale;
  const size_t i4 = (size_t)blockIdx.x * 256 + threadIdx.x;
  float4 v = *(const float4*)(w + i4 * 4);
  unsigned b0 = fp8_e4m3_from_small_int(fminf(fmaxf(rintf(v.x * inv), -1.0f), 1.0f));
  unsigned b1 = fp8_e4m3_from_small_int(fminf(fmaxf(rintf(v.y * inv), -1.0f), 1.0f));
  unsigned b2 = fp8_e4m3_from_small_int(fminf(fmaxf(rintf(v.z * inv), -1.0f), 1.0f));
  unsigned b3 = fp8_e4m3_from_small_int(fminf(fmaxf(rintf(v.w * inv), -1.0f), 1.0f));
  ((unsigned*)wq)[i4] = b0 | (b1 << 8) | (b2 << 16) | (b3 << 24);
}

__global__ __launch_bounds__(256) void fwht_quant_kernel(
    const float* __restrict__ x, const float* __restrict__ hs,
    const float* __restrict__ hb, unsigned char* __restrict__ actq,
    float* __restrict__ aScale) {
  __shared__ float row[IN_F];
  __shared__ float red[256];
  const int t = threadIdx.x;
  const size_t r = blockIdx.x;
  const float* xr = x + r * IN_F;
  for (int i = t; i < IN_F; i += 256) row[i] = xr[i];
  __syncthreads();
  for (int h = 1; h < IN_F; h <<= 1) {
#pragma unroll
    for (int pp = 0; pp < 8; ++pp) {
      int p = t + pp * 256;
      int j = ((p & ~(h - 1)) << 1) | (p & (h - 1));
      float a = row[j], b = row[j + h];
      row[j] = a + b;
      row[j + h] = a - b;
    }
    __syncthreads();
  }
  const float inv = 1.0f / 64.0f;
  const int base = t * 16;
  float vals[16];
  float mx = 0.0f;
#pragma unroll
  for (int i = 0; i < 16; ++i) {
    int idx = base + i;
    float v = row[idx] * inv * hs[idx] + hb[idx];
    vals[i] = v;
    mx = fmaxf(mx, fabsf(v));
  }
  red[t] = mx;
  __syncthreads();
  for (int off = 128; off > 0; off >>= 1) {
    if (t < off) red[t] = fmaxf(red[t], red[t + off]);
    __syncthreads();
  }
  const float gamma = red[0] + EPSF;
  const float qs = ACT_QB / gamma;
  union { unsigned char c[16]; int4 v; } u;
#pragma unroll
  for (int i = 0; i < 16; ++i) {
    float q = fminf(fmaxf(rintf(vals[i] * qs), -ACT_QB), ACT_QB);
    u.c[i] = (unsigned char)fp8_e4m3_from_small_int(q);
  }
  *(int4*)(actq + r * IN_F + base) = u.v;
  if (t == 0) aScale[r] = gamma * (1.0f / ACT_QB);
}

#define A_PITCH 136
#define B_PITCH 144
#define A_TILE  (64 * A_PITCH)
#define B_TILE  (128 * B_PITCH)
#define SMEM_BYTES (2 * (A_TILE + B_TILE))

__device__ __forceinline__ void issue_tile_copies(
    const unsigned char* __restrict__ Aq, const unsigned char* __restrict__ Bq,
    char* smem, unsigned aOff, unsigned bOff, int mBase, int nBase, int k, int t) {
  const size_t K = IN_F;
#pragma unroll
  for (int j = 0; j < 4; ++j) {
    int id = t * 4 + j;
    int row = id >> 4, col = (id & 15) * 8;
    async_b64(smem + aOff + row * A_PITCH + col,
              Aq + (size_t)(mBase + row) * K + k + col);
  }
#pragma unroll
  for (int j = 0; j < 4; ++j) {
    int id = t * 4 + j;
    int row = id >> 3, col = (id & 7) * 16;
    async_b128(smem + bOff + row * B_PITCH + col,
               Bq + (size_t)(nBase + row) * K + k + col);
  }
}

__global__ __launch_bounds__(256) void gemm_fp8_kernel(
    const unsigned char* __restrict__ Aq, const unsigned char* __restrict__ Bq,
    const float* __restrict__ aScale, const float* __restrict__ wsum,
    float* __restrict__ out) {
  extern __shared__ char smem[];
  const int t      = threadIdx.x;
  const int lane   = t & 31;
  const int wave   = t >> 5;
  const int laneLo = lane & 15;
  const int laneHi = lane >> 4;
  const int wm = wave & 1;
  const int wn = wave >> 1;
  const int mBase = blockIdx.y * 64;
  const int nBase = blockIdx.x * 128;
  const int m0 = mBase + wm * 32;
  const int n0 = nBase + wn * 32;

  const unsigned aOff[2] = {0u, A_TILE};
  const unsigned bOff[2] = {2u * A_TILE, 2u * A_TILE + B_TILE};

  const v8f vzero = {0.f, 0.f, 0.f, 0.f, 0.f, 0.f, 0.f, 0.f};
  v8f acc[2][2];
#pragma unroll
  for (int i = 0; i < 2; ++i)
#pragma unroll
    for (int j = 0; j < 2; ++j) acc[i][j] = vzero;

  issue_tile_copies(Aq, Bq, smem, aOff[0], bOff[0], mBase, nBase, 0, t);

#pragma unroll 2
  for (int kk = 0; kk < IN_F / 128; ++kk) {
    const int p = kk & 1;
    if (kk + 1 < IN_F / 128) {
      issue_tile_copies(Aq, Bq, smem, aOff[p ^ 1], bOff[p ^ 1],
                        mBase, nBase, (kk + 1) * 128, t);
      wait_async_le8();
    } else {
      wait_async_le0();
    }
    __syncthreads();

    v16i afrag[2];
#pragma unroll
    for (int mf = 0; mf < 2; ++mf) {
      const char* ap = smem + aOff[p] +
                       (wm * 32 + mf * 16 + laneLo) * A_PITCH + laneHi * 16;
      const int2* p2 = (const int2*)ap;
      int2 c0 = p2[0], c1 = p2[1], c2 = p2[4], c3 = p2[5];
      int2 c4 = p2[8], c5 = p2[9], c6 = p2[12], c7 = p2[13];
      v16i a = {c0.x, c0.y, c1.x, c1.y, c2.x, c2.y, c3.x, c3.y,
                c4.x, c4.y, c5.x, c5.y, c6.x, c6.y, c7.x, c7.y};
      afrag[mf] = a;
    }
    v16i bfrag[2];
#pragma unroll
    for (int nf = 0; nf < 2; ++nf) {
      const char* bp = smem + bOff[p] +
                       (wn * 32 + nf * 16 + laneLo) * B_PITCH + laneHi * 16;
      const int4* p4 = (const int4*)bp;
      int4 d0 = p4[0], d1 = p4[2], d2 = p4[4], d3 = p4[6];
      v16i b = {d0.x, d0.y, d0.z, d0.w, d1.x, d1.y, d1.z, d1.w,
                d2.x, d2.y, d2.z, d2.w, d3.x, d3.y, d3.z, d3.w};
      bfrag[nf] = b;
    }
#pragma unroll
    for (int nf = 0; nf < 2; ++nf) {
      acc[0][nf] = __builtin_amdgcn_wmma_f32_16x16x128_fp8_fp8(
          afrag[0], bfrag[nf], (short)0, acc[0][nf], false, false);
      acc[1][nf] = __builtin_amdgcn_wmma_f32_16x16x128_fp8_fp8(
          afrag[1], bfrag[nf], (short)0, acc[1][nf], false, false);
    }
    __syncthreads();
  }

  const float wScale = wsum[0] * (1.0f / ((float)IN_F * (float)OUT_F)) + EPSF;
#pragma unroll
  for (int mf = 0; mf < 2; ++mf) {
#pragma unroll
    for (int rr = 0; rr < 8; ++rr) {
      const int m = m0 + mf * 16 + laneHi * 8 + rr;
      const float sA = aScale[m] * wScale;
#pragma unroll
      for (int nf = 0; nf < 2; ++nf) {
        const int n = n0 + nf * 16 + laneLo;
        out[(size_t)m * OUT_F + n] = acc[mf][nf][rr] * sA;
      }
    }
  }
}

extern "C" void kernel_launch(void* const* d_in, const int* in_sizes, int n_in,
                              void* d_out, int out_size, void* d_ws, size_t ws_size,
                              hipStream_t stream) {
  (void)n_in; (void)out_size; (void)ws_size;
  const float* x  = (const float*)d_in[0];
  const float* w  = (const float*)d_in[1];
  const float* hs = (const float*)d_in[2];
  const float* hb = (const float*)d_in[3];
  float* out = (float*)d_out;
  const int M = in_sizes[0] / IN_F;

  char* ws = (char*)d_ws;
  float* wAbsSum      = (float*)ws;
  float* partials     = (float*)(ws + 256);
  float* aScale       = (float*)(ws + 16 * 1024);
  unsigned char* actq = (unsigned char*)(ws + 64 * 1024);
  unsigned char* wq   = actq + (size_t)M * IN_F;

  const int n4 = (OUT_F * IN_F) / 4;
  wabs_partial_kernel<<<RED_BLOCKS, 256, 0, stream>>>(w, partials, n4);
  wabs_final_kernel<<<1, 256, 0, stream>>>(partials, wAbsSum);
  wquant_kernel<<<n4 / 256, 256, 0, stream>>>(w, wAbsSum, wq);
  fwht_quant_kernel<<<M, 256, 0, stream>>>(x, hs, hb, actq, aScale);
  gemm_fp8_kernel<<<dim3(OUT_F / 128, M / 64), 256, SMEM_BYTES, stream>>>(
      actq, wq, aScale, wAbsSum, out);
}
